// CrossAttention_3186865733698
// MI455X (gfx1250) — compile-verified, matrix operand order checked by reading, not run on hardware
//
#include <hip/hip_runtime.h>

#define HEADS    16
#define DIM_HEAD 64
#define TLEN     16
#define BATCH    2048
#define DMODEL   1024
#define MROWS    (BATCH * TLEN)   /* 32768 */
#define NQKV     (3 * DMODEL)     /* 3072  */
#define MAXREL   16

typedef __attribute__((ext_vector_type(16))) __bf16 v16bf;
typedef __attribute__((ext_vector_type(8)))  __bf16 v8bf;
typedef __attribute__((ext_vector_type(8)))  float  v8f;
typedef __attribute__((ext_vector_type(4)))  int    v4i;

#if defined(__has_builtin)
#if __has_builtin(__builtin_amdgcn_global_load_async_to_lds_b128)
#define HAVE_ASYNC_LDS 1
#else
#define HAVE_ASYNC_LDS 0
#endif
#else
#define HAVE_ASYNC_LDS 0
#endif

static __device__ __forceinline__ unsigned short f2bf(float f) {
  union { float f; unsigned u; } v; v.f = f;
  unsigned r = v.u + 0x7FFFu + ((v.u >> 16) & 1u);   // round-to-nearest-even
  return (unsigned short)(r >> 16);
}
static __device__ __forceinline__ float bf2f(unsigned short u) {
  union { unsigned u; float f; } v; v.u = ((unsigned)u) << 16;
  return v.f;
}

// 16-byte global -> LDS copy: async on CDNA5 (GLOBAL_LOAD_ASYNC_TO_LDS_B128,
// tracked by ASYNCcnt), synchronous fallback otherwise.
static __device__ __forceinline__ void copy16_g2l(const unsigned short* g,
                                                  unsigned short* l) {
#if HAVE_ASYNC_LDS
  __builtin_amdgcn_global_load_async_to_lds_b128(
      (__attribute__((address_space(1))) v4i*)(g),
      (__attribute__((address_space(3))) v4i*)(l), 0, 0);
#else
  *(v8bf*)l = *(const v8bf*)g;
#endif
}
static __device__ __forceinline__ void async_wait0() {
#if HAVE_ASYNC_LDS
#if __has_builtin(__builtin_amdgcn_s_wait_asynccnt)
  __builtin_amdgcn_s_wait_asynccnt(0);
#else
  asm volatile("s_wait_asynccnt 0x0" ::: "memory");
#endif
#endif
}

// Build one 16x32 bf16 fragment (CDNA5 WMMA A layout; also used for the B
// operand of A*B^T, which is symmetric) from a row-major bf16 buffer with
// leading dimension `ld` (elements).
// Lane L<16 : row M=L,    halves hold K = 0..7  and 16..23.
// Lane L>=16: row M=L-16, halves hold K = 8..15 and 24..31.
static __device__ __forceinline__ v16bf load_frag(const unsigned short* p, int ld) {
  const int lane = threadIdx.x & 31;
  const unsigned short* q = p + (size_t)(lane & 15) * ld + ((lane < 16) ? 0 : 8);
  v8bf lo = *(const v8bf*)(q);
  v8bf hi = *(const v8bf*)(q + 16);
  v16bf r;
#pragma unroll
  for (int i = 0; i < 8; ++i) { r[i] = lo[i]; r[i + 8] = hi[i]; }
  return r;
}

// ---------------------------------------------------------------------------
// f32 -> bf16 conversion (grid-stride)
// ---------------------------------------------------------------------------
__global__ void cvt_f32_bf16(const float* __restrict__ src,
                             unsigned short* __restrict__ dst, size_t n) {
  size_t i = (size_t)blockIdx.x * blockDim.x + threadIdx.x;
  size_t stride = (size_t)gridDim.x * blockDim.x;
  for (; i < n; i += stride) dst[i] = f2bf(src[i]);
}

// ---------------------------------------------------------------------------
// C = A * B^T  (A: [M,K] bf16 row-major, B: [N,K] bf16 row-major)
// Block: 256 thr = 8 waves. Block tile 128(M) x 64(N), wave tile 32x32.
// Double-buffered LDS staging (K-slab = 64) fed by async global->LDS copies;
// fragments consumed via ds_load_b128. 72-elem row stride => conflict-free.
// BF16OUT ? bf16 C : f32 C + bias[n].
// ---------------------------------------------------------------------------
template <bool BF16OUT>
__global__ void __launch_bounds__(256)
gemm_abt(const unsigned short* __restrict__ A,
         const unsigned short* __restrict__ B,
         void* __restrict__ Cout, const float* __restrict__ bias,
         int M, int N, int K) {
  constexpr int BM = 128, BN = 64, BK = 64, LDT = BK + 8;  // 72
  __shared__ unsigned short As[2][BM * LDT];               // 2 x 18 KB
  __shared__ unsigned short Bs[2][BN * LDT];               // 2 x  9 KB

  const int tid  = threadIdx.x;
  const int wave = tid >> 5;
  const int wm   = (wave & 3) * 32;        // wave row offset in block tile
  const int wn   = (wave >> 2) * 32;       // wave col offset in block tile
  const int bm   = blockIdx.x * BM;
  const int bn   = blockIdx.y * BN;

  // Stage one K-slab (BK cols) of the A and B block tiles into LDS buffer.
  auto stage = [&](int buf, int k0) {
#pragma unroll
    for (int i = 0; i < 4; ++i) {          // A: 128 rows x 8 chunks = 1024
      int c = tid + i * 256;
      int row = c >> 3, col = (c & 7) * 8;
      copy16_g2l(A + (size_t)(bm + row) * K + k0 + col,
                 &As[buf][row * LDT + col]);
    }
#pragma unroll
    for (int i = 0; i < 2; ++i) {          // B: 64 rows x 8 chunks = 512
      int c = tid + i * 256;
      int row = c >> 3, col = (c & 7) * 8;
      copy16_g2l(B + (size_t)(bn + row) * K + k0 + col,
                 &Bs[buf][row * LDT + col]);
    }
  };

  stage(0, 0);
  async_wait0();
  __syncthreads();

  v8f c00 = {}, c01 = {}, c10 = {}, c11 = {};
  const int nsteps = K / BK;
  int buf = 0;
  for (int s = 0; s < nsteps; ++s) {
    if (s + 1 < nsteps) stage(buf ^ 1, (s + 1) * BK);  // prefetch next slab
#pragma unroll
    for (int kb = 0; kb < BK; kb += 32) {
      v16bf a0 = load_frag(&As[buf][(wm)      * LDT + kb], LDT);
      v16bf a1 = load_frag(&As[buf][(wm + 16) * LDT + kb], LDT);
      v16bf b0 = load_frag(&Bs[buf][(wn)      * LDT + kb], LDT);
      v16bf b1 = load_frag(&Bs[buf][(wn + 16) * LDT + kb], LDT);
      c00 = __builtin_amdgcn_wmma_f32_16x16x32_bf16(false, a0, false, b0, (short)0, c00, false, false);
      c01 = __builtin_amdgcn_wmma_f32_16x16x32_bf16(false, a0, false, b1, (short)0, c01, false, false);
      c10 = __builtin_amdgcn_wmma_f32_16x16x32_bf16(false, a1, false, b0, (short)0, c10, false, false);
      c11 = __builtin_amdgcn_wmma_f32_16x16x32_bf16(false, a1, false, b1, (short)0, c11, false, false);
    }
    if (s + 1 < nsteps) {
      async_wait0();
      __syncthreads();
      buf ^= 1;
    }
  }

  // C fragment layout: lane<16 -> rows r+0, lane>=16 -> rows r+8; col = lane&15
  const int lane = tid & 31;
  const int n0 = lane & 15;
  const int rb = (lane >> 4) << 3;
  const int gm0 = bm + wm, gn0 = bn + wn;
#pragma unroll
  for (int r = 0; r < 8; ++r) {
    const int m0 = rb + r;
    if (BF16OUT) {
      unsigned short* C = (unsigned short*)Cout;
      C[(size_t)(gm0 + m0)      * N + (gn0 + n0)]      = f2bf(c00[r]);
      C[(size_t)(gm0 + m0)      * N + (gn0 + 16 + n0)] = f2bf(c01[r]);
      C[(size_t)(gm0 + 16 + m0) * N + (gn0 + n0)]      = f2bf(c10[r]);
      C[(size_t)(gm0 + 16 + m0) * N + (gn0 + 16 + n0)] = f2bf(c11[r]);
    } else {
      float* C = (float*)Cout;
      C[(size_t)(gm0 + m0)      * N + (gn0 + n0)]      = c00[r] + bias[gn0 + n0];
      C[(size_t)(gm0 + m0)      * N + (gn0 + 16 + n0)] = c01[r] + bias[gn0 + 16 + n0];
      C[(size_t)(gm0 + 16 + m0) * N + (gn0 + n0)]      = c10[r] + bias[gn0 + n0];
      C[(size_t)(gm0 + 16 + m0) * N + (gn0 + 16 + n0)] = c11[r] + bias[gn0 + 16 + n0];
    }
  }
}

// ---------------------------------------------------------------------------
// Attention core: one wave per (batch, head). sim = q k^T via 2 WMMA (K=64),
// rel-k bias + softmax in the C-fragment layout with wave32 shuffles,
// attn*(v + rel_v) via VALU (16x16x64 per head -> tiny), bf16 output.
// QKV layout: [MROWS, 3072] with cols [0,1024)=Q, [1024,2048)=K, [2048,3072)=V.
// ---------------------------------------------------------------------------
__global__ void __launch_bounds__(256)
attn_kernel(const unsigned short* __restrict__ QKV,
            const float* __restrict__ relk,   // [2*MAXREL+1, 64]
            const float* __restrict__ relv,   // [2*MAXREL+1, 64]
            unsigned short* __restrict__ O) { // [MROWS, 1024] bf16
  __shared__ float attn_s[8][TLEN * 17];
  const int wave = threadIdx.x >> 5;
  const int lane = threadIdx.x & 31;
  const int head = blockIdx.x * 8 + wave;     // 0 .. BATCH*HEADS-1
  const int b = head >> 4;
  const int h = head & 15;

  const size_t base = (size_t)b * TLEN * NQKV + (size_t)h * DIM_HEAD;
  const unsigned short* qp = QKV + base;
  const unsigned short* kp = QKV + base + DMODEL;
  const unsigned short* vp = QKV + base + 2 * DMODEL;

  // sim = q @ k^T : one 16x16 tile, K = 64 -> two 16x16x32 WMMA
  v8f c = {};
  {
    v16bf a0 = load_frag(qp + 0,  NQKV);
    v16bf b0 = load_frag(kp + 0,  NQKV);
    c = __builtin_amdgcn_wmma_f32_16x16x32_bf16(false, a0, false, b0, (short)0, c, false, false);
    v16bf a1 = load_frag(qp + 32, NQKV);
    v16bf b1 = load_frag(kp + 32, NQKV);
    c = __builtin_amdgcn_wmma_f32_16x16x32_bf16(false, a1, false, b1, (short)0, c, false, false);
  }

  const float scale = 0.125f;                 // 1/sqrt(64)
  const int s  = lane & 15;                   // key position (column)
  const int rb = (lane >> 4) << 3;            // row base per lane half
#pragma unroll
  for (int r = 0; r < 8; ++r) {
    const int t = rb + r;                     // query position (row)
    int rel = s - t;
    rel = rel < -MAXREL ? -MAXREL : (rel > MAXREL ? MAXREL : rel);
    const float* krow = relk + (rel + MAXREL) * DIM_HEAD;
    const unsigned short* qrow = qp + (size_t)t * NQKV;
    float dot = 0.f;
#pragma unroll 8
    for (int d = 0; d < DIM_HEAD; ++d) dot += bf2f(qrow[d]) * krow[d];
    float val = (c[r] + dot) * scale;

    // row-wise softmax: the 16 lanes of each half-wave hold one row
    float mx = val;
#pragma unroll
    for (int off = 8; off >= 1; off >>= 1) mx = fmaxf(mx, __shfl_xor(mx, off, 32));
    float e = __expf(val - mx);
    float sm = e;
#pragma unroll
    for (int off = 8; off >= 1; off >>= 1) sm += __shfl_xor(sm, off, 32);
    attn_s[wave][t * 17 + s] = e / sm;
  }
  __syncthreads();

  // out[t,d] = sum_s attn[t,s] * (v[s,d] + rel_v[clip(s-t)+L, d])
  for (int i = lane; i < TLEN * DIM_HEAD; i += 32) {
    const int t = i >> 6;
    const int d = i & 63;
    float acc = 0.f;
#pragma unroll
    for (int ss = 0; ss < TLEN; ++ss) {
      int rel = ss - t;
      rel = rel < -MAXREL ? -MAXREL : (rel > MAXREL ? MAXREL : rel);
      const float a = attn_s[wave][t * 17 + ss];
      acc += a * (bf2f(vp[(size_t)ss * NQKV + d]) + relv[(rel + MAXREL) * DIM_HEAD + d]);
    }
    O[((size_t)b * TLEN + t) * DMODEL + (size_t)h * DIM_HEAD + d] = f2bf(acc);
  }
}

// ---------------------------------------------------------------------------
// Host launcher
// ---------------------------------------------------------------------------
extern "C" void kernel_launch(void* const* d_in, const int* in_sizes, int n_in,
                              void* d_out, int out_size, void* d_ws, size_t ws_size,
                              hipStream_t stream) {
  (void)in_sizes; (void)n_in; (void)out_size; (void)ws_size;
  const float* x    = (const float*)d_in[0];
  const float* Wq   = (const float*)d_in[1];
  const float* Wk   = (const float*)d_in[2];
  const float* Wv   = (const float*)d_in[3];
  const float* Wo   = (const float*)d_in[4];
  const float* bo   = (const float*)d_in[5];
  const float* relk = (const float*)d_in[6];
  const float* relv = (const float*)d_in[7];

  char* p = (char*)d_ws;
  auto carve = [&](size_t bytes) -> char* {
    char* r = p; p += (bytes + 255) & ~(size_t)255; return r;
  };
  unsigned short* xb   = (unsigned short*)carve((size_t)MROWS * DMODEL * 2);   //  64 MB
  unsigned short* Wqkv = (unsigned short*)carve((size_t)NQKV  * DMODEL * 2);   //   6 MB
  unsigned short* Wob  = (unsigned short*)carve((size_t)DMODEL * DMODEL * 2);  //   2 MB
  unsigned short* QKVb = (unsigned short*)carve((size_t)MROWS * NQKV * 2);     // 192 MB
  unsigned short* Ob   = (unsigned short*)carve((size_t)MROWS * DMODEL * 2);   //  64 MB

  const size_t nW = (size_t)DMODEL * DMODEL;
  cvt_f32_bf16<<<2048, 256, 0, stream>>>(x,  xb,   (size_t)MROWS * DMODEL);
  cvt_f32_bf16<<<256,  256, 0, stream>>>(Wq, Wqkv,          nW);
  cvt_f32_bf16<<<256,  256, 0, stream>>>(Wk, Wqkv + nW,     nW);
  cvt_f32_bf16<<<256,  256, 0, stream>>>(Wv, Wqkv + 2 * nW, nW);
  cvt_f32_bf16<<<256,  256, 0, stream>>>(Wo, Wob,           nW);

  // QKV = x @ [Wq;Wk;Wv]^T   (32768 x 3072 x 1024)
  gemm_abt<true><<<dim3(MROWS / 128, NQKV / 64), 256, 0, stream>>>(
      xb, Wqkv, QKVb, nullptr, MROWS, NQKV, DMODEL);

  // per-head attention (32768 heads, 8 waves/block)
  attn_kernel<<<(BATCH * HEADS) / 8, 256, 0, stream>>>(QKVb, relk, relv, Ob);

  // out = O @ Wo^T + bo      (32768 x 1024 x 1024, f32 out)
  gemm_abt<false><<<dim3(MROWS / 128, DMODEL / 64), 256, 0, stream>>>(
      Ob, Wob, d_out, bo, MROWS, DMODEL, DMODEL);
}
